// GAT_27376121545091
// MI455X (gfx1250) — matrix operand order repaired and checked by reading, not compiled, not run on hardware
//
#include <hip/hip_runtime.h>

#define N_NODES   20000
#define N_EDGES   320000
#define IN_FEATS  256
#define H_FEATS   64
#define NUM_HEADS 4
#define NEG_SLOPE 0.2f
#define ENC_NEG_INF 0x007FFFFFu   // enc(-inf) under order-preserving float->uint map

typedef __attribute__((ext_vector_type(16))) __bf16 v16bf;
typedef __attribute__((ext_vector_type(8)))  float  v8f;

// ---------- helpers ----------
static __device__ __forceinline__ __bf16 f2bf(float f) {
  union { float f; unsigned u; } in; in.f = f;
  unsigned u = in.u;
  unsigned r = 0x7FFFu + ((u >> 16) & 1u);          // round-to-nearest-even
  unsigned short hs = (unsigned short)((u + r) >> 16);
  union { unsigned short s; __bf16 b; } out; out.s = hs;
  return out.b;
}

// Order-preserving float <-> uint mapping so segment-max can use atomicMax(u32).
static __device__ __forceinline__ unsigned enc_f32(float f) {
  union { float f; unsigned u; } c; c.f = f;
  return (c.u & 0x80000000u) ? ~c.u : (c.u | 0x80000000u);
}
static __device__ __forceinline__ float dec_f32(unsigned u) {
  unsigned v = (u & 0x80000000u) ? (u & 0x7FFFFFFFu) : ~u;
  union { unsigned u; float f; } c; c.u = v;
  return c.f;
}

// ---------- generic fill ----------
__global__ void fill_u32(unsigned* __restrict__ p, unsigned v, int n) {
  int i = blockIdx.x * blockDim.x + threadIdx.x;
  if (i < n) p[i] = v;
}

// ---------- one-time packs (fp32 -> bf16), done once so GEMM loop is load+WMMA only ----
__global__ void pack_bf16(const float* __restrict__ src, __bf16* __restrict__ dst, int n) {
  int i = blockIdx.x * blockDim.x + threadIdx.x;
  if (i < n) dst[i] = f2bf(src[i]);
}
// W[K,NC] row-major -> Wt[NC,K] bf16 (each output column becomes a contiguous K-run)
__global__ void pack_wT_bf16(const float* __restrict__ W, __bf16* __restrict__ Wt,
                             int K, int NC) {
  int i = blockIdx.x * blockDim.x + threadIdx.x;
  if (i >= K * NC) return;
  int k = i / NC, n = i % NC;
  Wt[(size_t)n * K + k] = f2bf(W[i]);
}

// ---------- WMMA GEMM: Y[M,NC] = X[M,K] @ W[K,NC] (X bf16 row-major, Wt bf16 [NC,K]) ----
// One block per 16-row panel; NC/16 waves, one 16x16 column tile each.
// A panel (16 x K bf16, K=256 -> 8KB) staged in LDS cooperatively, fragments read
// as b128 chunks per the ISA 7.12.2 16-bit A layout; B fragments are two 16-byte chunks
// per lane from Wt, 16 bf16 apart. f32 accumulation.
__global__ __launch_bounds__(512)
void gat_gemm_wmma(const __bf16* __restrict__ X, const __bf16* __restrict__ Wt,
                   float* __restrict__ Y, int K, int NC) {
  __shared__ __bf16 As[16 * IN_FEATS];          // K <= 256

  const int m0   = blockIdx.x * 16;
  const int tid  = threadIdx.x;
  const int lane = tid & 31;
  const int wid  = tid >> 5;
  const int half = lane >> 4;                   // 0 or 1
  const int l16  = lane & 15;
  const int n0   = wid * 16;

  // cooperative stage of contiguous 16*K bf16 A panel -> LDS (b128 chunks)
  {
    const uint4* xs = reinterpret_cast<const uint4*>(X + (size_t)m0 * K);
    uint4*       as = reinterpret_cast<uint4*>(As);
    const int nchunk = (16 * K) / 8;            // 8 bf16 per uint4
    for (int c = tid; c < nchunk; c += blockDim.x) as[c] = xs[c];
  }
  __syncthreads();

  const __bf16* wrow = Wt + (size_t)(n0 + l16) * K;   // this lane's B column
  v8f acc = {};

  for (int k0 = 0; k0 < K; k0 += 32) {
    union { v16bf bf; uint4 u[2]; } A, B;
    // A frag: lanes 0-15 hold K chunks {k0..k0+7, k0+16..k0+23}; lanes 16-31 offset +8
    const __bf16* ap = &As[l16 * K + k0 + 8 * half];
    A.u[0] = *reinterpret_cast<const uint4*>(ap);
    A.u[1] = *reinterpret_cast<const uint4*>(ap + 16);
    // B frag: lanes 0-15 hold K chunks {k0..k0+7, k0+16..k0+23} of column n0+l16; lanes 16-31 offset +8
    const __bf16* bp = wrow + k0 + 8 * half;
    B.u[0] = *reinterpret_cast<const uint4*>(bp);
    B.u[1] = *reinterpret_cast<const uint4*>(bp + 16);

    if (k0 + 32 < K)                              // global_prefetch_b8 next B tile
      __builtin_prefetch(wrow + k0 + 32, 0, 3);

    acc = __builtin_amdgcn_wmma_f32_16x16x32_bf16(
        /*neg_a=*/false, A.bf, /*neg_b=*/false, B.bf,
        /*c_mod=*/(short)0, acc, /*reuse_a=*/false, /*reuse_b=*/false);
  }
  // C/D layout: VGPR r, lane l -> M = r + 8*(l/16), N = l%16
#pragma unroll
  for (int r = 0; r < 8; ++r) {
    int m = m0 + r + 8 * half;
    Y[(size_t)m * NC + n0 + l16] = acc[r];
  }
}

// ---------- el/er attention projections: el[n,h] = dot(feat[n,h,:], al[h,:]) ----------
__global__ void node_dots(const float* __restrict__ feat,
                          const float* __restrict__ al, const float* __restrict__ ar,
                          float* __restrict__ el, float* __restrict__ er,
                          int N, int H, int O) {
  int i = blockIdx.x * blockDim.x + threadIdx.x;
  if (i >= N * H) return;
  int n = i / H, h = i % H;
  const float* f = feat + (size_t)n * H * O + (size_t)h * O;
  float sl = 0.f, sr = 0.f;
  for (int o = 0; o < O; ++o) {
    float v = f[o];
    sl += v * al[h * O + o];
    sr += v * ar[h * O + o];
  }
  el[i] = sl;
  er[i] = sr;
}

// ---------- edge score + leaky relu + segment max (atomic on encoded uint) ----------
__global__ void edge_score(const float* __restrict__ el, const float* __restrict__ er,
                           const int* __restrict__ src, const int* __restrict__ dst,
                           float* __restrict__ e_out, unsigned* __restrict__ menc,
                           int E, int H) {
  int i = blockIdx.x * blockDim.x + threadIdx.x;
  if (i >= E * H) return;
  int e = i / H, h = i % H;
  float v = el[src[e] * H + h] + er[dst[e] * H + h];
  v = (v > 0.f) ? v : NEG_SLOPE * v;
  e_out[i] = v;
  atomicMax(&menc[dst[e] * H + h], enc_f32(v));
}

// ---------- exp(e - max) + segment sum ----------
__global__ void edge_exp(const float* __restrict__ e_in, const unsigned* __restrict__ menc,
                         const int* __restrict__ dst,
                         float* __restrict__ ex_out, float* __restrict__ denom,
                         int E, int H) {
  int i = blockIdx.x * blockDim.x + threadIdx.x;
  if (i >= E * H) return;
  int e = i / H, h = i % H;
  float m  = dec_f32(menc[dst[e] * H + h]);
  float ex = __expf(e_in[i] - m);
  ex_out[i] = ex;
  atomicAdd(&denom[dst[e] * H + h], ex);
}

// ---------- weighted message aggregation: one wave per (edge, head) ----------
__global__ void edge_aggr(const float* __restrict__ ex, const float* __restrict__ denom,
                          const int* __restrict__ src, const int* __restrict__ dst,
                          const float* __restrict__ feat, float* __restrict__ acc,
                          int E, int H, int O) {
  int wid  = (blockIdx.x * blockDim.x + threadIdx.x) >> 5;
  int lane = threadIdx.x & 31;
  if (wid >= E * H) return;
  int e = wid / H, h = wid % H;
  int s = src[e], d = dst[e];
  float alpha = ex[wid] / denom[d * H + h];
  const float* f = feat + (size_t)s * H * O + (size_t)h * O;
  float*       a = acc  + (size_t)d * H * O + (size_t)h * O;
  for (int o = lane; o < O; o += 32)
    atomicAdd(&a[o], alpha * f[o]);
}

// ---------- bias + ELU -> bf16 (feeds GEMM-2 as its A matrix) ----------
__global__ void bias_elu_bf16(const float* __restrict__ acc, const float* __restrict__ b,
                              __bf16* __restrict__ out, int N, int C) {
  int i = blockIdx.x * blockDim.x + threadIdx.x;
  if (i >= N * C) return;
  float v = acc[i] + b[i % C];
  v = (v > 0.f) ? v : (__expf(v) - 1.f);
  out[i] = f2bf(v);
}

// ---------- final bias -> fp32 output ----------
__global__ void bias_f32(const float* __restrict__ acc, const float* __restrict__ b,
                         float* __restrict__ out, int N, int C) {
  int i = blockIdx.x * blockDim.x + threadIdx.x;
  if (i >= N * C) return;
  out[i] = acc[i] + b[i % C];
}

// ---------- launch ----------
extern "C" void kernel_launch(void* const* d_in, const int* in_sizes, int n_in,
                              void* d_out, int out_size, void* d_ws, size_t ws_size,
                              hipStream_t stream) {
  const float* in_feat = (const float*)d_in[0];
  const float* W1  = (const float*)d_in[1];
  const float* al1 = (const float*)d_in[2];
  const float* ar1 = (const float*)d_in[3];
  const float* b1  = (const float*)d_in[4];
  const float* W2  = (const float*)d_in[5];
  const float* al2 = (const float*)d_in[6];
  const float* ar2 = (const float*)d_in[7];
  const float* b2  = (const float*)d_in[8];
  const int*   src = (const int*)d_in[9];
  const int*   dst = (const int*)d_in[10];
  float* out = (float*)d_out;

  const int C1 = NUM_HEADS * H_FEATS;                  // 256
  const size_t NF   = (size_t)N_NODES * C1;            // 5.12M
  const size_t NH1  = (size_t)N_NODES * NUM_HEADS;     // 80K
  const size_t EH1  = (size_t)N_EDGES * NUM_HEADS;     // 1.28M

  // Workspace layout (layer 2 reuses every region; deps enforced by stream order)
  float*    feat  = (float*)d_ws;                 // [N,256] f1  / [N,64] f2 (fp32)
  float*    acc   = feat  + NF;                   // [N,256] agg1 / [N,64] agg2
  __bf16*   Xbf   = (__bf16*)(acc + NF);          // [N,256] bf16: in_feat pack, then ELU(h)
  __bf16*   Wt1   = Xbf + NF;                     // [256,256] bf16 W1^T
  __bf16*   Wt2   = Wt1 + (size_t)IN_FEATS * C1;  // [64,256]  bf16 W2^T
  float*    el    = (float*)(Wt2 + (size_t)C1 * H_FEATS);
  float*    er    = el    + NH1;                  // [N,H]
  unsigned* menc  = (unsigned*)(er + NH1);        // [N,H] encoded segment max
  float*    denom = (float*)(menc + NH1);         // [N,H]
  float*    ebuf  = denom + NH1;                  // [E,H] scores
  float*    exbuf = ebuf  + EH1;                  // [E,H] exp(scores - max)

  const int T = 256;
  auto nb = [](size_t n, int t) { return (unsigned)((n + t - 1) / t); };

  // -------- one-time bf16 packs --------
  pack_bf16 <<<nb(NF, T), T, 0, stream>>>(in_feat, Xbf, (int)NF);
  pack_wT_bf16<<<nb((size_t)IN_FEATS * C1, T), T, 0, stream>>>(W1, Wt1, IN_FEATS, C1);
  pack_wT_bf16<<<nb((size_t)C1 * H_FEATS, T), T, 0, stream>>>(W2, Wt2, C1, H_FEATS);

  // ================= Layer 1 (4 heads, 64 feats) =================
  fill_u32<<<nb(NH1, T), T, 0, stream>>>(menc, ENC_NEG_INF, (int)NH1);
  fill_u32<<<nb(NH1, T), T, 0, stream>>>((unsigned*)denom, 0u, (int)NH1);
  fill_u32<<<nb(NF,  T), T, 0, stream>>>((unsigned*)acc,   0u, (int)NF);

  gat_gemm_wmma<<<N_NODES / 16, 32 * (C1 / 16), 0, stream>>>(Xbf, Wt1, feat, IN_FEATS, C1);

  node_dots<<<nb(NH1, T), T, 0, stream>>>(feat, al1, ar1, el, er, N_NODES, NUM_HEADS, H_FEATS);
  edge_score<<<nb(EH1, T), T, 0, stream>>>(el, er, src, dst, ebuf, menc, N_EDGES, NUM_HEADS);
  edge_exp<<<nb(EH1, T), T, 0, stream>>>(ebuf, menc, dst, exbuf, denom, N_EDGES, NUM_HEADS);
  edge_aggr<<<nb(EH1 * 32, T), T, 0, stream>>>(exbuf, denom, src, dst, feat, acc,
                                               N_EDGES, NUM_HEADS, H_FEATS);
  bias_elu_bf16<<<nb(NF, T), T, 0, stream>>>(acc, b1, Xbf, N_NODES, C1);  // h -> bf16 A

  // ================= Layer 2 (1 head, 64 feats) =================
  const size_t NH2 = (size_t)N_NODES;
  const size_t EH2 = (size_t)N_EDGES;
  const size_t NF2 = (size_t)N_NODES * H_FEATS;

  fill_u32<<<nb(NH2, T), T, 0, stream>>>(menc, ENC_NEG_INF, (int)NH2);
  fill_u32<<<nb(NH2, T), T, 0, stream>>>((unsigned*)denom, 0u, (int)NH2);
  fill_u32<<<nb(NF2, T), T, 0, stream>>>((unsigned*)acc,   0u, (int)NF2);

  gat_gemm_wmma<<<N_NODES / 16, 32 * (H_FEATS / 16), 0, stream>>>(Xbf, Wt2, feat, C1, H_FEATS);

  node_dots<<<nb(NH2, T), T, 0, stream>>>(feat, al2, ar2, el, er, N_NODES, 1, H_FEATS);
  edge_score<<<nb(EH2, T), T, 0, stream>>>(el, er, src, dst, ebuf, menc, N_EDGES, 1);
  edge_exp<<<nb(EH2, T), T, 0, stream>>>(ebuf, menc, dst, exbuf, denom, N_EDGES, 1);
  edge_aggr<<<nb(EH2 * 32, T), T, 0, stream>>>(exbuf, denom, src, dst, feat, acc,
                                               N_EDGES, 1, H_FEATS);
  bias_f32<<<nb(NF2, T), T, 0, stream>>>(acc, b2, out, N_NODES, H_FEATS);
}
